// Attention_73924977098811
// MI455X (gfx1250) — matrix operand order repaired and checked by reading, not compiled, not run on hardware
//
#include <hip/hip_runtime.h>
#include <hip/hip_bf16.h>
#include <stdint.h>

// ---------------------------------------------------------------------------
// CDNA5 (gfx1250) fused MHA: bf16 WMMA (v_wmma_f32_16x16x32_bf16) everywhere,
// f32 accumulate, flash-attention single pass, V pre-transposed per head.
// Round 2: replace manual RNE bit-twiddled f32->bf16 with native __bf16
// converts (hardware RNE, lowers to v_cvt_*bf16* instead of mov/and/add3
// chains) to cut VALU pressure on the staging and softmax->P paths.
// ---------------------------------------------------------------------------

typedef __bf16        v16bf __attribute__((ext_vector_type(16)));
typedef float         v8f   __attribute__((ext_vector_type(8)));
typedef unsigned int  v4u   __attribute__((ext_vector_type(4)));

union FragBF {
  v16bf v;
  v4u   q[2];
  unsigned short s[16];
};

#define D_MODEL 1024
#define HEADS   16
#define DH      64
#define SEQ     2048
#define BATCH   2

// Native hardware convert (RNE on CDNA5) instead of manual bit math.
__device__ __forceinline__ unsigned short f2bf(float f) {
  union { __bf16 h; unsigned short u; } c;
  c.h = (__bf16)f;
  return c.u;
}

// Paired convert so the compiler can fuse into packed bf16 converts.
__device__ __forceinline__ unsigned int pack2bf(float lo, float hi) {
  union { __bf16 h[2]; unsigned int u; } c;
  c.h[0] = (__bf16)lo;
  c.h[1] = (__bf16)hi;
  return c.u;
}

__device__ __forceinline__ v8f wmma_bf16(const FragBF& a, const FragBF& b, v8f c) {
  // 8 args: (neg_a, A, neg_b, B, c_mod, C, reuse_a, reuse_b)
  return __builtin_amdgcn_wmma_f32_16x16x32_bf16(false, a.v, false, b.v,
                                                 (short)0, c, false, false);
}

__device__ __forceinline__ void zero8(v8f& z) {
#pragma unroll
  for (int i = 0; i < 8; ++i) z[i] = 0.0f;
}

// ---------------------------------------------------------------------------
// Stage 0: f32 -> bf16 weight conversion (runs once per call, 4x 1M elements)
// ---------------------------------------------------------------------------
__global__ __launch_bounds__(256)
void cvt_f32_to_bf16(const float* __restrict__ src, unsigned short* __restrict__ dst, int n) {
  int i = (blockIdx.x * 256 + threadIdx.x) * 4;
  if (i + 3 < n) {
    float4 f = *(const float4*)(src + i);
    uint2 o;
    o.x = pack2bf(f.x, f.y);
    o.y = pack2bf(f.z, f.w);
    *(uint2*)(dst + i) = o;
  }
}

// ---------------------------------------------------------------------------
// Stage 1: QKV projection.  Y = X @ W^T + b, output bf16 in per-head layout.
// Grid (32 Mtiles, 16 Ntiles, 3 proj), 256 threads = 8 waves.
// Workgroup tile: 128 rows x 64 cols. Wave = 32x32 (2x2 WMMA tiles).
// ---------------------------------------------------------------------------
__global__ __launch_bounds__(256)
void qkv_proj_kernel(const float* __restrict__ Xq, const float* __restrict__ Xk,
                     const float* __restrict__ Xv,
                     const unsigned short* __restrict__ Wqkv,   // bf16 [3][D][D]
                     const float* __restrict__ Bq, const float* __restrict__ Bk,
                     const float* __restrict__ Bv,
                     unsigned short* __restrict__ Qb,           // [B][H][S][DH]
                     unsigned short* __restrict__ Kb,           // [B][H][S][DH]
                     unsigned short* __restrict__ Vt)           // [B][H][DH][S]
{
  __shared__ unsigned short ldsX[128 * 40];   // 128 rows x 32 cols bf16, stride 40

  const int mt = blockIdx.x;
  const int nt = blockIdx.y;
  const int pr = blockIdx.z;

  const float* X            = (pr == 0) ? Xq : (pr == 1) ? Xk : Xv;
  const float* bias         = (pr == 0) ? Bq : (pr == 1) ? Bk : Bv;
  const unsigned short* W   = Wqkv + (size_t)pr * D_MODEL * D_MODEL;

  const int tid    = threadIdx.x;
  const int lid    = tid & 31;
  const int wave   = tid >> 5;
  const int wm     = wave >> 1;      // 0..3 : 32-row strip
  const int wn     = wave & 1;       // 0..1 : 32-col strip
  const int lane16 = lid & 15;
  const int half   = lid >> 4;

  v8f acc[2][2];
  zero8(acc[0][0]); zero8(acc[0][1]); zero8(acc[1][0]); zero8(acc[1][1]);

  const int srow = tid >> 1;                 // staging: 2 threads per row
  const int scol = (tid & 1) << 4;
  const float* Xrow = X + (size_t)(mt * 128 + srow) * D_MODEL + scol;

  const int ncol0 = nt * 64 + wn * 32 + lane16;
  const unsigned short* Wr0 = W + (size_t)ncol0 * D_MODEL + half * 8;
  const unsigned short* Wr1 = W + (size_t)(ncol0 + 16) * D_MODEL + half * 8;
  const unsigned short* A0base = &ldsX[(wm * 32 + lane16) * 40];
  const unsigned short* A1base = &ldsX[(wm * 32 + 16 + lane16) * 40];

  for (int k0 = 0; k0 < D_MODEL; k0 += 32) {
    // ---- stage X[128][32] f32 -> bf16 into LDS (native packed converts) ----
    const float4* gp = (const float4*)(Xrow + k0);
    float4 f0 = gp[0], f1 = gp[1], f2 = gp[2], f3 = gp[3];
    unsigned int* lp = (unsigned int*)&ldsX[srow * 40 + scol];
    lp[0] = pack2bf(f0.x, f0.y);
    lp[1] = pack2bf(f0.z, f0.w);
    lp[2] = pack2bf(f1.x, f1.y);
    lp[3] = pack2bf(f1.z, f1.w);
    lp[4] = pack2bf(f2.x, f2.y);
    lp[5] = pack2bf(f2.z, f2.w);
    lp[6] = pack2bf(f3.x, f3.y);
    lp[7] = pack2bf(f3.z, f3.w);
    __syncthreads();

    // ---- A fragments (16x32 bf16) from LDS, B fragments (32x16) from L2 ----
    FragBF a0, a1, b0, b1;
    a0.q[0] = *(const v4u*)(A0base + half * 8);
    a0.q[1] = *(const v4u*)(A0base + 16 + half * 8);
    a1.q[0] = *(const v4u*)(A1base + half * 8);
    a1.q[1] = *(const v4u*)(A1base + 16 + half * 8);
    b0.q[0] = *(const v4u*)(Wr0 + k0);
    b0.q[1] = *(const v4u*)(Wr0 + k0 + 16);
    b1.q[0] = *(const v4u*)(Wr1 + k0);
    b1.q[1] = *(const v4u*)(Wr1 + k0 + 16);
    if (k0 + 64 < D_MODEL) __builtin_prefetch(Wr0 + k0 + 64, 0, 1);

    acc[0][0] = wmma_bf16(a0, b0, acc[0][0]);
    acc[0][1] = wmma_bf16(a0, b1, acc[0][1]);
    acc[1][0] = wmma_bf16(a1, b0, acc[1][0]);
    acc[1][1] = wmma_bf16(a1, b1, acc[1][1]);
    __syncthreads();
  }

  // ---- bias + store into per-head bf16 layouts ----
#pragma unroll
  for (int sv = 0; sv < 2; ++sv) {
    const int nc    = ncol0 + sv * 16;
    const float bval = bias[nc];
    const int h  = nc >> 6;
    const int dh = nc & 63;
#pragma unroll
    for (int su = 0; su < 2; ++su) {
#pragma unroll
      for (int i = 0; i < 8; ++i) {
        const int m  = mt * 128 + wm * 32 + su * 16 + half * 8 + i;
        const int bb = m >> 11;               // batch
        const int ss = m & (SEQ - 1);         // sequence pos
        const unsigned short val = f2bf(acc[su][sv][i] + bval);
        if (pr == 2)      Vt[(((size_t)bb * HEADS + h) * DH + dh) * SEQ + ss] = val;
        else if (pr == 1) Kb[(((size_t)bb * HEADS + h) * SEQ + ss) * DH + dh] = val;
        else              Qb[(((size_t)bb * HEADS + h) * SEQ + ss) * DH + dh] = val;
      }
    }
  }
}

// ---------------------------------------------------------------------------
// Stage 2: flash attention.  Grid (S/128, B*H), 256 threads = 8 waves.
// Each wave: 16 Q rows, loops over 32-key tiles: 4 WMMA QK^T + online softmax
// + 4 WMMA PV.  No 1/sqrt(Dh) scaling (matches reference).
// ---------------------------------------------------------------------------
__global__ __launch_bounds__(256)
void attn_kernel(const unsigned short* __restrict__ Qb,
                 const unsigned short* __restrict__ Kb,
                 const unsigned short* __restrict__ Vt,
                 const unsigned char* __restrict__ Mask,   // [B][1][S][S] bool
                 unsigned short* __restrict__ Xo)          // [B][S][D] bf16
{
  __shared__ unsigned short ldsP[8 * 16 * 40];   // per-wave 16x32 P tile, stride 40

  const int bh = blockIdx.y;
  const int b  = bh >> 4;
  const int h  = bh & 15;
  const int tid = threadIdx.x, lid = tid & 31, wave = tid >> 5;
  const int lane16 = lid & 15, half = lid >> 4;
  const int mr = blockIdx.x * 128 + wave * 16;   // Q row base of this wave

  const unsigned short* Qh = Qb + ((size_t)b * HEADS + h) * SEQ * DH;
  const unsigned short* Kh = Kb + ((size_t)b * HEADS + h) * SEQ * DH;
  const unsigned short* Vh = Vt + ((size_t)b * HEADS + h) * DH * SEQ;
  const unsigned char*  Mb = Mask + (size_t)b * SEQ * SEQ;

  // Q A-fragments (16x32 each, dh 0..31 / 32..63) held in registers all loop.
  FragBF aq0, aq1;
  {
    const unsigned short* qrow = Qh + (size_t)(mr + lane16) * DH;
    aq0.q[0] = *(const v4u*)(qrow + half * 8);
    aq0.q[1] = *(const v4u*)(qrow + 16 + half * 8);
    aq1.q[0] = *(const v4u*)(qrow + 32 + half * 8);
    aq1.q[1] = *(const v4u*)(qrow + 48 + half * 8);
  }

  float rm[8], rl[8];
  v8f acc[4];
#pragma unroll
  for (int i = 0; i < 8; ++i) { rm[i] = -1e30f; rl[i] = 0.0f; }
#pragma unroll
  for (int t = 0; t < 4; ++t) zero8(acc[t]);

  unsigned short* Pl = &ldsP[wave * 16 * 40];
  const int qr0 = mr + half * 8;

  for (int kt = 0; kt < SEQ; kt += 32) {
    // ---- S = Q K^T for two 16-key subtiles ----
    v8f s[2];
#pragma unroll
    for (int t = 0; t < 2; ++t) {
      FragBF bk0, bk1;
      const unsigned short* krow = Kh + (size_t)(kt + t * 16 + lane16) * DH;
      bk0.q[0] = *(const v4u*)(krow + half * 8);
      bk0.q[1] = *(const v4u*)(krow + 16 + half * 8);
      bk1.q[0] = *(const v4u*)(krow + 32 + half * 8);
      bk1.q[1] = *(const v4u*)(krow + 48 + half * 8);
      v8f z; zero8(z);
      z = wmma_bf16(aq0, bk0, z);
      z = wmma_bf16(aq1, bk1, z);
      s[t] = z;
    }
    if (kt + 32 < SEQ) __builtin_prefetch(Kh + (size_t)(kt + 32) * DH, 0, 1);

    // ---- mask fill (-1e5 on True) ----
#pragma unroll
    for (int i = 0; i < 8; ++i) {
      const unsigned char* mrow = Mb + (size_t)(qr0 + i) * SEQ + kt + lane16;
      if (mrow[0])  s[0][i] = -100000.0f;
      if (mrow[16]) s[1][i] = -100000.0f;
    }

    // ---- online softmax (row = 16-lane group in C layout) ----
    float mnew[8], corr[8];
#pragma unroll
    for (int i = 0; i < 8; ++i) {
      float t = fmaxf(s[0][i], s[1][i]);
      t = fmaxf(t, __shfl_xor(t, 1, 32));
      t = fmaxf(t, __shfl_xor(t, 2, 32));
      t = fmaxf(t, __shfl_xor(t, 4, 32));
      t = fmaxf(t, __shfl_xor(t, 8, 32));
      mnew[i] = fmaxf(rm[i], t);
      corr[i] = __expf(rm[i] - mnew[i]);
      rm[i]   = mnew[i];
    }
#pragma unroll
    for (int i = 0; i < 8; ++i) {
      float p0 = __expf(s[0][i] - mnew[i]);
      float p1 = __expf(s[1][i] - mnew[i]);
      s[0][i] = p0; s[1][i] = p1;
      float ts = p0 + p1;
      ts += __shfl_xor(ts, 1, 32);
      ts += __shfl_xor(ts, 2, 32);
      ts += __shfl_xor(ts, 4, 32);
      ts += __shfl_xor(ts, 8, 32);
      rl[i] = rl[i] * corr[i] + ts;
    }
#pragma unroll
    for (int t = 0; t < 4; ++t)
#pragma unroll
      for (int i = 0; i < 8; ++i) acc[t][i] *= corr[i];

    // ---- P (16x32) -> LDS bf16, re-read as A fragment ----
#pragma unroll
    for (int i = 0; i < 8; ++i) {
      const int r = i + half * 8;
      Pl[r * 40 + lane16]      = f2bf(s[0][i]);
      Pl[r * 40 + 16 + lane16] = f2bf(s[1][i]);
    }
    asm volatile("s_wait_dscnt 0" ::: "memory");   // cross-lane LDS handoff in-wave

    FragBF ap;
    const unsigned short* prow = Pl + lane16 * 40;
    ap.q[0] = *(const v4u*)(prow + half * 8);
    ap.q[1] = *(const v4u*)(prow + 16 + half * 8);

    // ---- O += P V : B columns contiguous thanks to transposed V ----
#pragma unroll
    for (int t = 0; t < 4; ++t) {
      FragBF bv;
      const unsigned short* vrow = Vh + (size_t)(t * 16 + lane16) * SEQ + kt + half * 8;
      bv.q[0] = *(const v4u*)(vrow);
      bv.q[1] = *(const v4u*)(vrow + 16);
      acc[t] = wmma_bf16(ap, bv, acc[t]);
    }
  }

  // ---- normalize and store bf16 [B][S][H*DH] ----
  float inv[8];
#pragma unroll
  for (int i = 0; i < 8; ++i) inv[i] = 1.0f / rl[i];
#pragma unroll
  for (int t = 0; t < 4; ++t)
#pragma unroll
    for (int i = 0; i < 8; ++i) {
      const int srow = qr0 + i;
      const int dh   = t * 16 + lane16;
      Xo[((size_t)b * SEQ + srow) * D_MODEL + h * DH + dh] = f2bf(acc[t][i] * inv[i]);
    }
}

// ---------------------------------------------------------------------------
// Stage 3: output projection.  Out = Xo @ Wo^T + bo, f32 result.
// Grid (32, 16), 256 threads = 8 waves; A fragments direct from global bf16.
// ---------------------------------------------------------------------------
__global__ __launch_bounds__(256)
void out_proj_kernel(const unsigned short* __restrict__ Xo,
                     const unsigned short* __restrict__ Wo,   // bf16 [D][D]
                     const float* __restrict__ Bo,
                     float* __restrict__ Out)
{
  const int mt = blockIdx.x, nt = blockIdx.y;
  const int tid = threadIdx.x, lid = tid & 31, wave = tid >> 5;
  const int wm = wave >> 1, wn = wave & 1;
  const int lane16 = lid & 15, half = lid >> 4;

  const int m0 = mt * 128 + wm * 32 + lane16;
  const unsigned short* X0 = Xo + (size_t)m0 * D_MODEL;
  const unsigned short* X1 = Xo + (size_t)(m0 + 16) * D_MODEL;
  const int ncol0 = nt * 64 + wn * 32 + lane16;
  const unsigned short* Wr0 = Wo + (size_t)ncol0 * D_MODEL + half * 8;
  const unsigned short* Wr1 = Wo + (size_t)(ncol0 + 16) * D_MODEL + half * 8;

  v8f acc[2][2];
  zero8(acc[0][0]); zero8(acc[0][1]); zero8(acc[1][0]); zero8(acc[1][1]);

  for (int k0 = 0; k0 < D_MODEL; k0 += 32) {
    FragBF a0, a1, b0, b1;
    a0.q[0] = *(const v4u*)(X0 + k0 + half * 8);
    a0.q[1] = *(const v4u*)(X0 + k0 + 16 + half * 8);
    a1.q[0] = *(const v4u*)(X1 + k0 + half * 8);
    a1.q[1] = *(const v4u*)(X1 + k0 + 16 + half * 8);
    b0.q[0] = *(const v4u*)(Wr0 + k0);
    b0.q[1] = *(const v4u*)(Wr0 + k0 + 16);
    b1.q[0] = *(const v4u*)(Wr1 + k0);
    b1.q[1] = *(const v4u*)(Wr1 + k0 + 16);
    acc[0][0] = wmma_bf16(a0, b0, acc[0][0]);
    acc[0][1] = wmma_bf16(a0, b1, acc[0][1]);
    acc[1][0] = wmma_bf16(a1, b0, acc[1][0]);
    acc[1][1] = wmma_bf16(a1, b1, acc[1][1]);
  }

#pragma unroll
  for (int sv = 0; sv < 2; ++sv) {
    const int nc = ncol0 + sv * 16;
    const float bval = Bo[nc];
#pragma unroll
    for (int su = 0; su < 2; ++su) {
#pragma unroll
      for (int i = 0; i < 8; ++i) {
        const int m = mt * 128 + wm * 32 + su * 16 + half * 8 + i;
        Out[(size_t)m * D_MODEL + nc] = acc[su][sv][i] + bval;
      }
    }
  }
}

// ---------------------------------------------------------------------------
// Host launcher
// ---------------------------------------------------------------------------
extern "C" void kernel_launch(void* const* d_in, const int* in_sizes, int n_in,
                              void* d_out, int out_size, void* d_ws, size_t ws_size,
                              hipStream_t stream) {
  const float*         q    = (const float*)d_in[0];
  const float*         k    = (const float*)d_in[1];
  const float*         v    = (const float*)d_in[2];
  const unsigned char* mask = (const unsigned char*)d_in[3];
  const float* wq = (const float*)d_in[4];  const float* bq = (const float*)d_in[5];
  const float* wk = (const float*)d_in[6];  const float* bk = (const float*)d_in[7];
  const float* wv = (const float*)d_in[8];  const float* bv = (const float*)d_in[9];
  const float* wo = (const float*)d_in[10]; const float* bo = (const float*)d_in[11];
  float* out = (float*)d_out;

  // Workspace layout (bf16 as ushort), 40 MB total, rewritten every call:
  unsigned short* ws  = (unsigned short*)d_ws;
  const size_t WSZ = (size_t)D_MODEL * D_MODEL;           // 1,048,576
  const size_t TSZ = (size_t)BATCH * HEADS * SEQ * DH;    // 4,194,304
  unsigned short* WbQ = ws;
  unsigned short* WbK = WbQ + WSZ;
  unsigned short* WbV = WbK + WSZ;
  unsigned short* WbO = WbV + WSZ;
  unsigned short* Qb  = WbO + WSZ;
  unsigned short* Kb  = Qb + TSZ;
  unsigned short* Vt  = Kb + TSZ;
  unsigned short* Xo  = Vt + TSZ;

  // Stage 0: weights f32 -> bf16 (row-major: W^T columns contiguous)
  cvt_f32_to_bf16<<<1024, 256, 0, stream>>>(wq, WbQ, (int)WSZ);
  cvt_f32_to_bf16<<<1024, 256, 0, stream>>>(wk, WbK, (int)WSZ);
  cvt_f32_to_bf16<<<1024, 256, 0, stream>>>(wv, WbV, (int)WSZ);
  cvt_f32_to_bf16<<<1024, 256, 0, stream>>>(wo, WbO, (int)WSZ);

  // Stage 1: QKV projections -> per-head bf16 (V transposed)
  qkv_proj_kernel<<<dim3(32, 16, 3), 256, 0, stream>>>(q, k, v, WbQ, bq, bk, bv,
                                                       Qb, Kb, Vt);
  // Stage 2: flash attention
  attn_kernel<<<dim3(SEQ / 128, BATCH * HEADS), 256, 0, stream>>>(Qb, Kb, Vt, mask, Xo);

  // Stage 3: output projection -> f32
  out_proj_kernel<<<dim3(32, 16), 256, 0, stream>>>(Xo, WbO, bo, out);
}
